// PretrainedEmbedding_43147241456175
// MI455X (gfx1250) — compile-verified, matrix operand order checked by reading, not run on hardware
//
#include <hip/hip_runtime.h>

// ---------------------------------------------------------------------------
// Problem constants (from reference): B=64 S=256 F=4 V=50257 H=768 D=512
// tokens = B*S = 16384, 16 tokens per block -> 1024 blocks.
// ---------------------------------------------------------------------------
#define Hdim 768
#define Ddim 512
#define Fsub 4
#define KT   (Hdim / 32)   // 24 K-tiles of 32
#define NT   (Ddim / 16)   // 32 N-tiles of 16
#define LDS_STRIDE 776     // 768 + 8 halfs pad -> 388-dword row stride, conflict-free

typedef __attribute__((ext_vector_type(16))) _Float16 v16h;
typedef __attribute__((ext_vector_type(8)))  _Float16 v8h;
typedef __attribute__((ext_vector_type(4)))  _Float16 v4h;
typedef __attribute__((ext_vector_type(8)))  float    v8f;

// ---------------------------------------------------------------------------
// Pre-pack W[768][512] f32 -> f16 in WMMA B-fragment layout.
// Fragment (nt, kt): lane l supplies column n = nt*16 + (l&15); its 16 half
// elements are K = kt*32 + 16*(i/8) + 8*(l/16) + (i%8)   (ISA 7.12.2 16-bit).
// Stored so each lane's 16 halfs (32B) are contiguous, fragments contiguous in
// kt -> the GEMM inner loop reads 1KB fully-coalesced chunks out of L2.
// ---------------------------------------------------------------------------
__global__ __launch_bounds__(256) void pack_w_kernel(const float* __restrict__ W,
                                                     _Float16* __restrict__ Wp) {
    int gwave = (blockIdx.x * 256 + threadIdx.x) >> 5;
    int lane  = threadIdx.x & 31;
    if (gwave >= NT * KT) return;
    int nt = gwave / KT;
    int kt = gwave % KT;
    int n  = nt * 16 + (lane & 15);
    int kb = kt * 32 + ((lane >> 4) << 3);
    _Float16* dst = Wp + (((size_t)(nt * KT + kt) * 32) + lane) * 16;
#pragma unroll
    for (int i = 0; i < 16; ++i) {
        int k = kb + ((i >> 3) << 4) + (i & 7);
        dst[i] = (_Float16)W[(size_t)k * Ddim + n];
    }
}

// ---------------------------------------------------------------------------
// Fused: masked-mean embedding gather (f32 -> f16 in LDS) + WMMA GEMM + bias.
// Block = 256 threads = 8 waves; block bm owns tokens [bm*16, bm*16+16).
// ---------------------------------------------------------------------------
__global__ __launch_bounds__(256) void embed_gemm_kernel(const int*       __restrict__ words,
                                                         const float*     __restrict__ E,
                                                         const _Float16*  __restrict__ Wp,
                                                         const float*     __restrict__ bias,
                                                         float*           __restrict__ out) {
    __shared__ _Float16 Ash[16][LDS_STRIDE];

    const int bm  = blockIdx.x;      // 0..1023
    const int tid = threadIdx.x;

    // ---------------- Phase A: gather + masked mean -> LDS (f16) -----------
    {
        const int row = tid >> 4;            // token row 0..15
        const int sub = tid & 15;            // 16 threads cooperate per row
        const int tok = bm * 16 + row;
        const int* wptr = words + (size_t)tok * Fsub;
        const int id0 = wptr[0], id1 = wptr[1], id2 = wptr[2], id3 = wptr[3];
        const int cnt = (id0 != 0) + (id1 != 0) + (id2 != 0) + (id3 != 0);
        const float scale = (cnt > 0) ? (1.0f / (float)cnt) : 0.0f;
        const float4* e0 = (const float4*)(E + (size_t)id0 * Hdim);
        const float4* e1 = (const float4*)(E + (size_t)id1 * Hdim);
        const float4* e2 = (const float4*)(E + (size_t)id2 * Hdim);
        const float4* e3 = (const float4*)(E + (size_t)id3 * Hdim);
#pragma unroll
        for (int jj = 0; jj < (Hdim / 4) / 16; ++jj) {   // 12 float4 per thread
            const int vi = sub + jj * 16;                // coalesced across sub
            float ax = 0.f, ay = 0.f, az = 0.f, aw = 0.f;
            if (id0) { float4 v = e0[vi]; ax += v.x; ay += v.y; az += v.z; aw += v.w; }
            if (id1) { float4 v = e1[vi]; ax += v.x; ay += v.y; az += v.z; aw += v.w; }
            if (id2) { float4 v = e2[vi]; ax += v.x; ay += v.y; az += v.z; aw += v.w; }
            if (id3) { float4 v = e3[vi]; ax += v.x; ay += v.y; az += v.z; aw += v.w; }
            v4h h;
            h[0] = (_Float16)(ax * scale);
            h[1] = (_Float16)(ay * scale);
            h[2] = (_Float16)(az * scale);
            h[3] = (_Float16)(aw * scale);
            *(v4h*)(&Ash[row][vi * 4]) = h;              // 8B aligned store
        }
    }
    __syncthreads();

    // ---------------- Phase B: WMMA GEMM over K=768 ------------------------
    const int lane = tid & 31;
    const int wv   = tid >> 5;               // wave 0..7, owns 4 N-tiles
    const int aoff = (lane >> 4) << 3;       // 0 or 8 halfs (ISA A layout)
    const _Float16* arow = &Ash[lane & 15][0];

    v8f acc[4];
#pragma unroll
    for (int q = 0; q < 4; ++q) acc[q] = (v8f)0.0f;

    const _Float16* wp0[4];
#pragma unroll
    for (int q = 0; q < 4; ++q) {
        const int nt = wv * 4 + q;
        wp0[q] = Wp + (((size_t)nt * KT) * 32 + lane) * 16;
    }

    for (int kt = 0; kt < KT; ++kt) {
        const int kb = kt * 32;
        // A fragment: two conflict-free ds_load_b128 (16B each, 16B aligned)
        v8h lo = *(const v8h*)(arow + kb + aoff);
        v8h hi = *(const v8h*)(arow + kb + 16 + aoff);
        v16h a = __builtin_shufflevector(lo, hi,
                    0, 1, 2, 3, 4, 5, 6, 7, 8, 9, 10, 11, 12, 13, 14, 15);
        const size_t kstep = (size_t)kt * 32 * 16;
#pragma unroll
        for (int q = 0; q < 4; ++q) {
            v16h bfr = *(const v16h*)(wp0[q] + kstep);   // 32B coalesced from L2
            acc[q] = __builtin_amdgcn_wmma_f32_16x16x32_f16(
                         false, a, false, bfr, (short)0, acc[q], false, false);
        }
    }

    // ---------------- Store D + bias (ISA C/D layout) ----------------------
    const int mbase = (lane >> 4) * 8;
    const int ncol  = lane & 15;
#pragma unroll
    for (int q = 0; q < 4; ++q) {
        const int nt = wv * 4 + q;
        const int d  = nt * 16 + ncol;
        const float bv = bias[d];
#pragma unroll
        for (int j = 0; j < 8; ++j) {
            const int m = mbase + j;
            out[((size_t)(bm * 16 + m)) * Ddim + d] = acc[q][j] + bv;
        }
    }
}

// ---------------------------------------------------------------------------
extern "C" void kernel_launch(void* const* d_in, const int* in_sizes, int n_in,
                              void* d_out, int out_size, void* d_ws, size_t ws_size,
                              hipStream_t stream) {
    const int*   words = (const int*)d_in[0];    // [B,S,F] ids (int32)
    const float* E     = (const float*)d_in[1];  // [V,H]
    const float* W     = (const float*)d_in[2];  // [H,D]
    const float* b     = (const float*)d_in[3];  // [D]
    float*       out   = (float*)d_out;          // [B,S,D]
    _Float16*    Wp    = (_Float16*)d_ws;        // 768 KB packed W (f16)

    const int tokens = in_sizes[0] / Fsub;       // 16384
    const int blocks = tokens / 16;              // 1024

    const int packWaves = NT * KT;               // 768 fragments, 1 wave each
    pack_w_kernel<<<(packWaves * 32 + 255) / 256, 256, 0, stream>>>(W, Wp);
    embed_gemm_kernel<<<blocks, 256, 0, stream>>>(words, E, Wp, b, out);
}
